// HBitLinear_86208583566046
// MI455X (gfx1250) — matrix operand order repaired and checked by reading, not compiled, not run on hardware
//
#include <hip/hip_runtime.h>

// ---------------------------------------------------------------------------
// HBitLinear for MI455X (gfx1250, wave32, WMMA)
// fwht(fwht(x_q) @ W_q^T) == x_q @ (H W_q^T H): fold both FWHTs into W once,
// then one f16 WMMA GEMM with exact int4 activations and f32 row-scale epilogue.
// GEMM main loop uses double-buffered LDS (one barrier per K-tile) so global
// staging loads are covered by a full WMMA compute phase.
// ---------------------------------------------------------------------------

typedef __attribute__((ext_vector_type(16))) _Float16     v16h;
typedef __attribute__((ext_vector_type(8)))  float        v8f;
typedef __attribute__((ext_vector_type(4)))  unsigned int u32x4;

#define D     2048
#define ROWS  16384          // B * S = 4 * 4096
#define NW    (D * D)        // 4194304 weight elements

// ------------------------- K1a: per-block sum(|W|) -------------------------
__global__ __launch_bounds__(256) void k_wabs_partial(const float* __restrict__ W,
                                                      float* __restrict__ partials) {
  __shared__ float red[256];
  const int t = threadIdx.x;
  const float4* W4 = (const float4*)W;
  float4 v = W4[(size_t)blockIdx.x * 256 + t];
  red[t] = fabsf(v.x) + fabsf(v.y) + fabsf(v.z) + fabsf(v.w);
  __syncthreads();
  for (int off = 128; off > 0; off >>= 1) {
    if (t < off) red[t] += red[t + off];
    __syncthreads();
  }
  if (t == 0) partials[blockIdx.x] = red[0];
}

// ------------------- K1b: deterministic final reduction --------------------
__global__ __launch_bounds__(256) void k_wabs_final(const float* __restrict__ partials,
                                                    float* __restrict__ wsum) {
  __shared__ float red[256];
  const int t = threadIdx.x;
  float s = 0.f;
  for (int j = 0; j < 16; ++j) s += partials[t + j * 256];  // fixed order
  red[t] = s;
  __syncthreads();
  for (int off = 128; off > 0; off >>= 1) {
    if (t < off) red[t] += red[t + off];
    __syncthreads();
  }
  if (t == 0) wsum[0] = red[0];
}

// ---------- K2: ternarize W row o, FWHT along d, write transposed ----------
__global__ __launch_bounds__(256) void k_tern_fwht_rows(const float* __restrict__ W,
                                                        const float* __restrict__ wsum,
                                                        float* __restrict__ B1T) {
  __shared__ float buf[D];
  const int t = threadIdx.x;
  const int o = blockIdx.x;
  const float wscale = fmaxf(wsum[0] * (1.0f / (float)NW), 1e-6f);
  const float thr = 0.5f * wscale;
  for (int j = 0; j < 8; ++j) {
    const int i = t + j * 256;
    const float w = W[(size_t)o * D + i];
    buf[i] = (w > thr) ? 1.0f : ((w < -thr) ? -1.0f : 0.0f);
  }
  __syncthreads();
  for (int st = 0; st < 11; ++st) {           // 2048-point in-place FWHT
    const int h = 1 << st;
    for (int q = 0; q < 4; ++q) {
      const int p  = t + q * 256;             // pair id 0..1023, disjoint
      const int ia = ((p >> st) << (st + 1)) + (p & (h - 1));
      const float a = buf[ia], b = buf[ia + h];
      buf[ia]     = a + b;
      buf[ia + h] = a - b;
    }
    __syncthreads();
  }
  for (int j = 0; j < 8; ++j) {
    const int d = t + j * 256;
    B1T[(size_t)d * D + o] = buf[d];          // transposed scratch
  }
}

// ------- K3: FWHT along o, scale, pack f16 into WhT[n][k] (N-major) --------
__global__ __launch_bounds__(256) void k_fwht_cols_pack(const float* __restrict__ B1T,
                                                        const float* __restrict__ wsum,
                                                        _Float16* __restrict__ WhT) {
  __shared__ float buf[D];
  const int t = threadIdx.x;
  const int d = blockIdx.x;                   // K index
  const float wscale = fmaxf(wsum[0] * (1.0f / (float)NW), 1e-6f);
  for (int j = 0; j < 8; ++j) {
    const int o = t + j * 256;
    buf[o] = B1T[(size_t)d * D + o];          // contiguous reads
  }
  __syncthreads();
  for (int st = 0; st < 11; ++st) {
    const int h = 1 << st;
    for (int q = 0; q < 4; ++q) {
      const int p  = t + q * 256;
      const int ia = ((p >> st) << (st + 1)) + (p & (h - 1));
      const float a = buf[ia], b = buf[ia + h];
      buf[ia]     = a + b;
      buf[ia + h] = a - b;
    }
    __syncthreads();
  }
  for (int j = 0; j < 8; ++j) {
    const int o = t + j * 256;
    WhT[(size_t)o * D + d] = (_Float16)(buf[o] * wscale);  // W'[k=d][n=o] @ [n][k]
  }
}

// ------------- K4: fused LayerNorm + absmax + 4-bit quantize ---------------
__global__ __launch_bounds__(256) void k_ln_quant(const float* __restrict__ x,
                                                  const float* __restrict__ gamma,
                                                  const float* __restrict__ beta,
                                                  _Float16* __restrict__ Xh,
                                                  float* __restrict__ rowscale) {
  __shared__ float red[256];
  const int t = threadIdx.x;
  const size_t r = blockIdx.x;
  float xv[8];
  float s = 0.f;
  for (int j = 0; j < 8; ++j) {
    const int i = t + j * 256;
    xv[j] = x[r * D + i];
    s += xv[j];
  }
  red[t] = s; __syncthreads();
  for (int off = 128; off > 0; off >>= 1) { if (t < off) red[t] += red[t + off]; __syncthreads(); }
  const float mu = red[0] * (1.0f / (float)D);
  __syncthreads();

  float s2 = 0.f;
  for (int j = 0; j < 8; ++j) { const float dd = xv[j] - mu; s2 += dd * dd; }
  red[t] = s2; __syncthreads();
  for (int off = 128; off > 0; off >>= 1) { if (t < off) red[t] += red[t + off]; __syncthreads(); }
  const float rs = rsqrtf(red[0] * (1.0f / (float)D) + 1e-5f);
  __syncthreads();

  float xl[8];
  float am = 0.f;
  for (int j = 0; j < 8; ++j) {
    const int i = t + j * 256;
    xl[j] = (xv[j] - mu) * rs * gamma[i] + beta[i];
    am = fmaxf(am, fabsf(xl[j]));
  }
  red[t] = am; __syncthreads();
  for (int off = 128; off > 0; off >>= 1) { if (t < off) red[t] = fmaxf(red[t], red[t + off]); __syncthreads(); }
  const float scale = fmaxf(red[0], 1e-6f);
  const float inv = 7.0f / scale;
  for (int j = 0; j < 8; ++j) {
    const int i = t + j * 256;
    float q = rintf(xl[j] * inv);                 // RNE, matches jnp.round
    q = fminf(7.0f, fmaxf(-7.0f, q));
    Xh[r * D + i] = (_Float16)q;                  // exact small int in f16
  }
  if (t == 0) rowscale[r] = scale * (1.0f / 7.0f);
}

// --------------------------- K5: WMMA f16 GEMM -----------------------------
// out[m][n] = rowscale[m] * sum_k Xh[m][k] * WhT[n][k]
// 128x128 block tile, BK=32, 8 waves (wave32) as 4(M) x 2(N), wave tile 32x64.
// Double-buffered LDS: one barrier per K-tile, global loads covered by the
// full WMMA phase of the same iteration.
#define BM  128
#define BN  128
#define BK  32
#define LDK 40   // padded pitch in halves (80B rows -> conflict-free b128)

__global__ __launch_bounds__(256) void k_gemm(const _Float16* __restrict__ Xh,
                                              const _Float16* __restrict__ WhT,
                                              const float* __restrict__ rowscale,
                                              float* __restrict__ out) {
  __shared__ _Float16 As[2][BM * LDK];   // 2 x 10 KB
  __shared__ _Float16 Bs[2][BN * LDK];   // 2 x 10 KB   (40 KB total of 320 KB/WGP)

  const int tid  = threadIdx.x;
  const int lane = tid & 31;
  const int wid  = tid >> 5;       // 0..7 (wave32)
  const int wm   = wid >> 1;       // 0..3 (M)
  const int wn   = wid & 1;        // 0..1 (N)

  const int m0 = blockIdx.y * BM;
  const int n0 = blockIdx.x * BN;

  const int lrow = tid >> 1;          // 0..127
  const int lcol = (tid & 1) * 16;    // 0 or 16 halves

  const _Float16* gA = Xh  + (size_t)(m0 + lrow) * D + lcol;
  const _Float16* gB = WhT + (size_t)(n0 + lrow) * D + lcol;

  u32x4 ra0 = *(const u32x4*)(gA);
  u32x4 ra1 = *(const u32x4*)(gA + 8);
  u32x4 rb0 = *(const u32x4*)(gB);
  u32x4 rb1 = *(const u32x4*)(gB + 8);

  v8f acc[2][4] = {};

  // ISA 7.12.2 fragment addressing (16-bit, wave32):
  // A 16x32: lanes 0-15 -> K 0..7 & 16..23 ; lanes 16-31 -> K 8..15 & 24..31
  // B 32x16: lanes 0-15 -> K 0..7 & 16..23 ; lanes 16-31 -> K 8..15 & 24..31
  const int lm    = lane & 15;
  const int koffA = (lane < 16) ? 0 : 8;
  const int koffB = (lane < 16) ? 0 : 8;

  union HV { v16h v; u32x4 q[2]; };

  for (int kt = 0; kt < D / BK; ++kt) {
    _Float16* as = As[kt & 1];
    _Float16* bs = Bs[kt & 1];

    // Commit the tile prefetched during the previous iteration.
    *(u32x4*)(&as[lrow * LDK + lcol    ]) = ra0;
    *(u32x4*)(&as[lrow * LDK + lcol + 8]) = ra1;
    *(u32x4*)(&bs[lrow * LDK + lcol    ]) = rb0;
    *(u32x4*)(&bs[lrow * LDK + lcol + 8]) = rb1;
    __syncthreads();                         // only barrier per K-tile

    // Prefetch the next tile; waited only at next iteration's commit.
    if (kt + 1 < D / BK) {
      const _Float16* nA = gA + (size_t)(kt + 1) * BK;
      const _Float16* nB = gB + (size_t)(kt + 1) * BK;
      ra0 = *(const u32x4*)(nA);
      ra1 = *(const u32x4*)(nA + 8);
      rb0 = *(const u32x4*)(nB);
      rb1 = *(const u32x4*)(nB + 8);
    }

    HV a[2], b[4];
#pragma unroll
    for (int tm = 0; tm < 2; ++tm) {
      const _Float16* base = &as[(wm * 32 + tm * 16 + lm) * LDK];
      a[tm].q[0] = *(const u32x4*)(base + koffA);
      a[tm].q[1] = *(const u32x4*)(base + 16 + koffA);
    }
#pragma unroll
    for (int tn = 0; tn < 4; ++tn) {
      const _Float16* base = &bs[(wn * 64 + tn * 16 + lm) * LDK];
      b[tn].q[0] = *(const u32x4*)(base + koffB);
      b[tn].q[1] = *(const u32x4*)(base + 16 + koffB);
    }

#pragma unroll
    for (int tm = 0; tm < 2; ++tm)
#pragma unroll
      for (int tn = 0; tn < 4; ++tn)
        acc[tm][tn] = __builtin_amdgcn_wmma_f32_16x16x32_f16(
            false, a[tm].v, false, b[tn].v, (short)0, acc[tm][tn], false, false);
    // NOTE: no barrier here. Iteration kt+1 writes LDS[(kt+1)&1], whose last
    // readers were iteration kt-1; those reads completed before every wave
    // passed iteration kt's barrier.
  }

  // Epilogue: C layout -> VGPR e: lanes 0-15 M=e, lanes 16-31 M=8+e; N=lane%16.
#pragma unroll
  for (int tm = 0; tm < 2; ++tm) {
    const int mb = m0 + wm * 32 + tm * 16 + ((lane < 16) ? 0 : 8);
#pragma unroll
    for (int tn = 0; tn < 4; ++tn) {
      const int col = n0 + wn * 64 + tn * 16 + lm;
#pragma unroll
      for (int e = 0; e < 8; ++e) {
        const int row = mb + e;
        out[(size_t)row * D + col] = acc[tm][tn][e] * rowscale[row];
      }
    }
  }
}

// ---------------------------------------------------------------------------
extern "C" void kernel_launch(void* const* d_in, const int* in_sizes, int n_in,
                              void* d_out, int out_size, void* d_ws, size_t ws_size,
                              hipStream_t stream) {
  const float* x     = (const float*)d_in[0];
  const float* W     = (const float*)d_in[1];
  const float* gamma = (const float*)d_in[2];
  const float* beta  = (const float*)d_in[3];
  float* out = (float*)d_out;

  // Workspace layout (256B aligned), total ~88.2 MB.
  char* ws = (char*)d_ws;
  float*    partials = (float*)(ws);                                    // 16 KB
  float*    wsum     = (float*)(ws + (16u << 10));                      // 4 B
  float*    B1T      = (float*)(ws + (32u << 10));                      // 16 MB
  _Float16* WhT      = (_Float16*)(ws + (32u << 10) + (16u << 20));     // 8 MB
  _Float16* Xh       = (_Float16*)(ws + (32u << 10) + (24u << 20));     // 64 MB
  float*    rowscale = (float*)(ws + (32u << 10) + (88u << 20));        // 64 KB

  k_wabs_partial  <<<4096, 256, 0, stream>>>(W, partials);
  k_wabs_final    <<<1,    256, 0, stream>>>(partials, wsum);
  k_tern_fwht_rows<<<2048, 256, 0, stream>>>(W, wsum, B1T);
  k_fwht_cols_pack<<<2048, 256, 0, stream>>>(B1T, wsum, WhT);
  k_ln_quant      <<<ROWS, 256, 0, stream>>>(x, gamma, beta, Xh, rowscale);
  k_gemm          <<<dim3(D / BN, ROWS / BM), 256, 0, stream>>>(Xh, WhT, rowscale, out);
}
